// MultiScaleRetention_33200097199002
// MI455X (gfx1250) — matrix operand order repaired and checked by reading, not compiled, not run on hardware
//
#include <hip/hip_runtime.h>
#include <hip/hip_bf16.h>

// ---------------------------------------------------------------------------
// MultiScaleRetention on MI455X (gfx1250): compute-bound on matmuls
// (~380 GFLOP vs ~200MB traffic @23.3TB/s) -> route all GEMMs through
// v_wmma_f32_16x16x32_f16 (wave32, 16x16x32 tiles, fp32 accumulation).
// __launch_bounds__(256,1): 8 wave32 waves / WGP -> 2 per SIMD32, which
// unlocks the large VGPR budget and prevents accumulator spills to scratch.
// ---------------------------------------------------------------------------

typedef __attribute__((ext_vector_type(8)))  _Float16 v8h;
typedef __attribute__((ext_vector_type(16))) _Float16 v16h;
typedef __attribute__((ext_vector_type(8)))  float    v8f;

union AFrag { v16h v; v8h h[2]; };

__device__ __forceinline__ v8f wmma_f16(const v16h& a, const v16h& b, const v8f& c) {
  // (neg_a, A, neg_b, B, c_mod, C, reuse_a, reuse_b)
  return __builtin_amdgcn_wmma_f32_16x16x32_f16(false, a, false, b, (short)0, c, false, false);
}

// ---------------------------------------------------------------------------
// fp32 -> f16 conversion (grid-stride)
// ---------------------------------------------------------------------------
__global__ void __launch_bounds__(256, 1)
cvt_f32_to_f16(const float* __restrict__ src, _Float16* __restrict__ dst, int n) {
  for (int i = blockIdx.x * blockDim.x + threadIdx.x; i < n; i += gridDim.x * blockDim.x)
    dst[i] = (_Float16)src[i];
}

// ---------------------------------------------------------------------------
// WMMA GEMM: C(MxN, OutT) = A(MxK, f16) * B(KxN, f16)
// Block tile 128x128x32, 256 threads = 8 wave32 waves in a 2x4 grid,
// each wave owns a 64x32 sub-tile = 4x2 WMMA 16x16 accumulators.
// ---------------------------------------------------------------------------
template <typename OutT>
__global__ void __launch_bounds__(256, 1)
gemm_wmma(const _Float16* __restrict__ A, const _Float16* __restrict__ B,
          OutT* __restrict__ C, int M, int N, int K) {
  constexpr int BM = 128, BN = 128, BK = 32;
  __shared__ _Float16 As[BM * BK];   // [m][k]
  __shared__ _Float16 Bs[BN * BK];   // transposed: [n][k] so B-frags are K-contiguous

  const int tid  = threadIdx.x;
  const int lane = tid & 31;
  const int w    = tid >> 5;
  const int hl   = lane & 15;        // lane % 16 : row (A) / col (B,C)
  const int hi   = lane >> 4;        // lane half-select for K chunks
  const int m0 = blockIdx.y * BM, n0 = blockIdx.x * BN;
  const int waveM = (w >> 2) * 64;   // 2 wave rows
  const int waveN = (w & 3) * 32;    // 4 wave cols

  v8f acc[4][2];
#pragma unroll
  for (int i = 0; i < 4; ++i)
#pragma unroll
    for (int j = 0; j < 2; ++j)
#pragma unroll
      for (int e = 0; e < 8; ++e) acc[i][j][e] = 0.f;

  for (int kk = 0; kk < K; kk += BK) {
    // stage A tile 128x32 (cvt fp32->f16 in flight)
#pragma unroll
    for (int i = tid; i < BM * (BK / 4); i += 256) {
      int r = i >> 3;              // BK/4 = 8 float4 per row
      int c = (i & 7) << 2;
      float4 f = *(const float4*)(A + (size_t)(m0 + r) * K + kk + c);
      _Float16* dp = As + r * BK + c;
      dp[0] = (_Float16)f.x; dp[1] = (_Float16)f.y;
      dp[2] = (_Float16)f.z; dp[3] = (_Float16)f.w;
    }
    // stage B tile 32x128 transposed -> Bs[n][k]
#pragma unroll
    for (int i = tid; i < BK * (BN / 4); i += 256) {
      int r = i >> 5;              // k row
      int c = (i & 31) << 2;       // n
      float4 f = *(const float4*)(B + (size_t)(kk + r) * N + n0 + c);
      Bs[(c + 0) * BK + r] = (_Float16)f.x;
      Bs[(c + 1) * BK + r] = (_Float16)f.y;
      Bs[(c + 2) * BK + r] = (_Float16)f.z;
      Bs[(c + 3) * BK + r] = (_Float16)f.w;
    }
    if (kk + BK < K) {  // gfx1250 global_prefetch_b8 of the next tiles
      __builtin_prefetch(A + (size_t)(m0 + (tid >> 1)) * K + kk + BK, 0, 1);
      __builtin_prefetch(B + (size_t)(kk + BK + (tid >> 4)) * N + n0, 0, 1);
    }
    __syncthreads();

    AFrag a[4], b[2];
#pragma unroll
    for (int mt = 0; mt < 4; ++mt) {
      int row = waveM + mt * 16 + hl;
      a[mt].h[0] = *(const v8h*)(As + row * BK + hi * 8);        // K 0..7 / 8..15
      a[mt].h[1] = *(const v8h*)(As + row * BK + 16 + hi * 8);   // K 16..23 / 24..31
    }
#pragma unroll
    for (int nt = 0; nt < 2; ++nt) {
      int col = waveN + nt * 16 + hl;
      b[nt].h[0] = *(const v8h*)(Bs + col * BK + hi * 8);        // K 0..7 / 8..15
      b[nt].h[1] = *(const v8h*)(Bs + col * BK + 16 + hi * 8);
    }
#pragma unroll
    for (int mt = 0; mt < 4; ++mt)
#pragma unroll
      for (int nt = 0; nt < 2; ++nt)
        acc[mt][nt] = wmma_f16(a[mt].v, b[nt].v, acc[mt][nt]);
    __syncthreads();
  }

  // C/D layout: lane 0-15 -> M=r, lane 16-31 -> M=r+8; N = lane%16
#pragma unroll
  for (int mt = 0; mt < 4; ++mt)
#pragma unroll
    for (int nt = 0; nt < 2; ++nt) {
      int row = m0 + waveM + mt * 16 + hi * 8;
      int col = n0 + waveN + nt * 16 + hl;
#pragma unroll
      for (int r = 0; r < 8; ++r)
        C[(size_t)(row + r) * N + col] = (OutT)acc[mt][nt][r];
    }
}

// ---------------------------------------------------------------------------
// RoPE in-place on f16 tensor [rows=B*T*H][256]; scale folds DK^-0.5 for q.
// ---------------------------------------------------------------------------
__global__ void __launch_bounds__(256, 1)
rope_half_inplace(_Float16* __restrict__ x, int T_, int rows, float scale) {
  int gid = blockIdx.x * blockDim.x + threadIdx.x;
  if (gid >= rows * 128) return;
  int i   = gid & 127;           // rotary pair index, d/2 = 128
  int row = gid >> 7;            // (b*T + t)*H + h
  int t   = (row >> 3) & (T_ - 1);  // H = 8, T power of two
  // inv_freq = 10000^(-2i/256) = exp2(-(i/128)*log2(10000))
  float inv = exp2f(-(float)i * (13.287712379549449f / 128.f));
  float s, c;
  __sincosf((float)t * inv, &s, &c);
  _Float16* p = x + (size_t)row * 256;
  float x1 = (float)p[i], x2 = (float)p[i + 128];
  p[i]       = (_Float16)((x1 * c - x2 * s) * scale);
  p[i + 128] = (_Float16)((x2 * c + x1 * s) * scale);
}

// ---------------------------------------------------------------------------
// Retention: per (b, h, 64-row query block) stream causal 64-row key blocks.
//   S = Qblk @ Kblk^T      (WMMA, f32 accum)  -> decay mask -> f16 via LDS
//   O += S @ Vblk          (WMMA, f32 accum, V staged transposed)
// 8 waves: S phase each wave does 2 of 16 S-tiles; O phase each wave owns
// a 64-wide DV slice (4x4 tiles). Dynamic LDS = 136 KB (WGP has 320 KB).
// ---------------------------------------------------------------------------
__global__ void __launch_bounds__(256, 1)
retention_wmma(const _Float16* __restrict__ q, const _Float16* __restrict__ k,
               const _Float16* __restrict__ v, float* __restrict__ o, int T_) {
  constexpr int H_ = 8, DK_ = 256, DV_ = 512, QB = 64, KB = 64;
  extern __shared__ char smem[];
  _Float16* Qs = (_Float16*)smem;       // [64][256]
  _Float16* Ks = Qs + QB * DK_;         // [64][256]
  _Float16* Vt = Ks + KB * DK_;         // [512][64] transposed: [dv][krow]
  _Float16* Ss = Vt + DV_ * KB;         // [64][64]

  const int tid = threadIdx.x, lane = tid & 31, w = tid >> 5;
  const int hl = lane & 15, hi = lane >> 4;
  const int i0 = blockIdx.x * QB;
  const int h  = blockIdx.y;
  const int b  = blockIdx.z;
  const float log2g = log2f(1.f - exp2f(-5.f - (float)h));

  const size_t qk_stride = (size_t)H_ * DK_;   // per-t stride
  const size_t v_stride  = (size_t)H_ * DV_;
  const _Float16* qbase = q + ((size_t)(b * T_) * H_ + h) * DK_;
  const _Float16* kbase = k + ((size_t)(b * T_) * H_ + h) * DK_;
  const _Float16* vbase = v + ((size_t)(b * T_) * H_ + h) * DV_;

  // stage Q block (already RoPE'd + scaled)
  for (int idx = tid; idx < QB * DK_ / 8; idx += 256) {
    int r = idx >> 5, c = (idx & 31) << 3;
    *(v8h*)(Qs + r * DK_ + c) = *(const v8h*)(qbase + (size_t)(i0 + r) * qk_stride + c);
  }

  v8f accO[4][4];
#pragma unroll
  for (int i = 0; i < 4; ++i)
#pragma unroll
    for (int j = 0; j < 4; ++j)
#pragma unroll
      for (int e = 0; e < 8; ++e) accO[i][j][e] = 0.f;
  __syncthreads();

  for (int j0 = 0; j0 <= i0; j0 += KB) {
    // stage K block
    for (int idx = tid; idx < KB * DK_ / 8; idx += 256) {
      int r = idx >> 5, c = (idx & 31) << 3;
      *(v8h*)(Ks + r * DK_ + c) = *(const v8h*)(kbase + (size_t)(j0 + r) * qk_stride + c);
    }
    // stage V block transposed -> Vt[dv][krow]
    for (int idx = tid; idx < KB * DV_ / 8; idx += 256) {
      int r = idx >> 6, c = (idx & 63) << 3;
      v8h val = *(const v8h*)(vbase + (size_t)(j0 + r) * v_stride + c);
#pragma unroll
      for (int e = 0; e < 8; ++e) Vt[(c + e) * KB + r] = val[e];
    }
    __syncthreads();

    // ---- S = Q K^T : each wave computes 2 of the 16 16x16 tiles ----
#pragma unroll
    for (int tt = 0; tt < 2; ++tt) {
      int t = w * 2 + tt, tm = t >> 2, tn = t & 3;
      v8f s;
#pragma unroll
      for (int e = 0; e < 8; ++e) s[e] = 0.f;
      int row = tm * 16 + hl;
      int col = tn * 16 + hl;
#pragma unroll
      for (int kk = 0; kk < DK_; kk += 32) {
        AFrag a, bb;
        a.h[0]  = *(const v8h*)(Qs + row * DK_ + kk + hi * 8);
        a.h[1]  = *(const v8h*)(Qs + row * DK_ + kk + 16 + hi * 8);
        bb.h[0] = *(const v8h*)(Ks + col * DK_ + kk + hi * 8);
        bb.h[1] = *(const v8h*)(Ks + col * DK_ + kk + 16 + hi * 8);
        s = wmma_f16(a.v, bb.v, s);
      }
      // causal decay mask in C-layout, then f16 to LDS for the A-layout reload
      int m_base = tm * 16 + hi * 8;
#pragma unroll
      for (int r = 0; r < 8; ++r) {
        int diff = (i0 + m_base + r) - (j0 + col);
        float dec = (diff >= 0) ? exp2f(log2g * (float)diff) : 0.f;
        Ss[(m_base + r) * KB + col] = (_Float16)(s[r] * dec);
      }
    }
    __syncthreads();

    // ---- O += S @ V : wave w owns dv in [w*64, w*64+64) ----
#pragma unroll
    for (int tm = 0; tm < 4; ++tm) {
      int row = tm * 16 + hl;
      AFrag a0, a1;  // contraction over 64 key rows = 2 WMMA K-steps
      a0.h[0] = *(const v8h*)(Ss + row * KB + hi * 8);
      a0.h[1] = *(const v8h*)(Ss + row * KB + 16 + hi * 8);
      a1.h[0] = *(const v8h*)(Ss + row * KB + 32 + hi * 8);
      a1.h[1] = *(const v8h*)(Ss + row * KB + 48 + hi * 8);
#pragma unroll
      for (int tn = 0; tn < 4; ++tn) {
        int dv = w * 64 + tn * 16 + hl;
        AFrag b0, b1;
        b0.h[0] = *(const v8h*)(Vt + dv * KB + hi * 8);
        b0.h[1] = *(const v8h*)(Vt + dv * KB + 16 + hi * 8);
        b1.h[0] = *(const v8h*)(Vt + dv * KB + 32 + hi * 8);
        b1.h[1] = *(const v8h*)(Vt + dv * KB + 48 + hi * 8);
        accO[tm][tn] = wmma_f16(a0.v, b0.v, accO[tm][tn]);
        accO[tm][tn] = wmma_f16(a1.v, b1.v, accO[tm][tn]);
      }
    }
    __syncthreads();
  }

  float* obase = o + ((size_t)(b * T_) * H_ + h) * DV_;
#pragma unroll
  for (int tm = 0; tm < 4; ++tm)
#pragma unroll
    for (int tn = 0; tn < 4; ++tn) {
      int dv = w * 64 + tn * 16 + hl;
      int m_base = tm * 16 + hi * 8;
#pragma unroll
      for (int r = 0; r < 8; ++r)
        obase[(size_t)(i0 + m_base + r) * v_stride + dv] = accO[tm][tn][r];
    }
}

// ---------------------------------------------------------------------------
// RMS-norm over DV=512 + SiLU gate; one wave32 per (b,t,h) row.
// ---------------------------------------------------------------------------
__global__ void __launch_bounds__(256, 1)
norm_gate(const float* __restrict__ o, const _Float16* __restrict__ g,
          const float* __restrict__ gw, _Float16* __restrict__ out, int rows) {
  int w = threadIdx.x >> 5, lane = threadIdx.x & 31;
  int row = blockIdx.x * 8 + w;
  if (row >= rows) return;
  const float* op = o + (size_t)row * 512;
  float vals[16], ss = 0.f;
#pragma unroll
  for (int j = 0; j < 16; ++j) { vals[j] = op[lane + j * 32]; ss += vals[j] * vals[j]; }
#pragma unroll
  for (int off = 16; off > 0; off >>= 1) ss += __shfl_xor(ss, off, 32);
  float rms = rsqrtf(ss * (1.f / 512.f) + 1e-5f);
  const _Float16* gp = g + (size_t)row * 512;
  _Float16* dp = out + (size_t)row * 512;
#pragma unroll
  for (int j = 0; j < 16; ++j) {
    int dv = lane + j * 32;
    float gvf = (float)gp[dv];
    float gate = gvf / (1.f + __expf(-gvf));           // silu(g)
    dp[dv] = (_Float16)(vals[j] * rms * gw[dv] * gate);
  }
}

// ---------------------------------------------------------------------------
// Host launcher
// ---------------------------------------------------------------------------
extern "C" void kernel_launch(void* const* d_in, const int* in_sizes, int n_in,
                              void* d_out, int out_size, void* d_ws, size_t ws_size,
                              hipStream_t stream) {
  (void)in_sizes; (void)n_in; (void)out_size; (void)ws_size;
  constexpr int B = 2, T = 2048, HID = 2048, H = 8, DK = 256, DV = 512;
  constexpr int M = B * T;              // 4096 token rows
  constexpr int NQK = H * DK;           // 2048
  constexpr int NVG = H * DV;           // 4096

  const float* hs  = (const float*)d_in[0];
  const float* Wq  = (const float*)d_in[1];
  const float* Wk  = (const float*)d_in[2];
  const float* Wv  = (const float*)d_in[3];
  const float* Wg  = (const float*)d_in[4];
  const float* Wo  = (const float*)d_in[5];
  const float* gnw = (const float*)d_in[6];
  float* out = (float*)d_out;

  char* ws = (char*)d_ws;
  size_t off = 0;
  auto alloc = [&](size_t bytes) -> void* {
    void* p = ws + off;
    off = (off + bytes + 255) & ~(size_t)255;
    return p;
  };
  _Float16* hs_h  = (_Float16*)alloc((size_t)M * HID * 2);
  _Float16* Wq_h  = (_Float16*)alloc((size_t)HID * NQK * 2);
  _Float16* Wk_h  = (_Float16*)alloc((size_t)HID * NQK * 2);
  _Float16* Wv_h  = (_Float16*)alloc((size_t)HID * NVG * 2);
  _Float16* Wg_h  = (_Float16*)alloc((size_t)HID * NVG * 2);
  _Float16* Wo_h  = (_Float16*)alloc((size_t)NVG * HID * 2);
  _Float16* q_h   = (_Float16*)alloc((size_t)M * NQK * 2);
  _Float16* k_h   = (_Float16*)alloc((size_t)M * NQK * 2);
  _Float16* v_h   = (_Float16*)alloc((size_t)M * NVG * 2);
  _Float16* g_h   = (_Float16*)alloc((size_t)M * NVG * 2);
  float*    o_f32 = (float*)   alloc((size_t)M * NVG * 4);
  _Float16* gat_h = (_Float16*)alloc((size_t)M * NVG * 2);

  // 1) fp32 -> f16 conversions
  cvt_f32_to_f16<<<4096, 256, 0, stream>>>(hs, hs_h, M * HID);
  cvt_f32_to_f16<<<4096, 256, 0, stream>>>(Wq, Wq_h, HID * NQK);
  cvt_f32_to_f16<<<4096, 256, 0, stream>>>(Wk, Wk_h, HID * NQK);
  cvt_f32_to_f16<<<4096, 256, 0, stream>>>(Wv, Wv_h, HID * NVG);
  cvt_f32_to_f16<<<4096, 256, 0, stream>>>(Wg, Wg_h, HID * NVG);
  cvt_f32_to_f16<<<4096, 256, 0, stream>>>(Wo, Wo_h, NVG * HID);

  // 2) projections (WMMA GEMMs, f16 out)
  gemm_wmma<_Float16><<<dim3(NQK / 128, M / 128), 256, 0, stream>>>(hs_h, Wq_h, q_h, M, NQK, HID);
  gemm_wmma<_Float16><<<dim3(NQK / 128, M / 128), 256, 0, stream>>>(hs_h, Wk_h, k_h, M, NQK, HID);
  gemm_wmma<_Float16><<<dim3(NVG / 128, M / 128), 256, 0, stream>>>(hs_h, Wv_h, v_h, M, NVG, HID);
  gemm_wmma<_Float16><<<dim3(NVG / 128, M / 128), 256, 0, stream>>>(hs_h, Wg_h, g_h, M, NVG, HID);

  // 3) RoPE (q gets DK^-0.5 folded in: 256^-0.5 = 0.0625)
  rope_half_inplace<<<(M * H * 128) / 256, 256, 0, stream>>>(q_h, T, M * H, 0.0625f);
  rope_half_inplace<<<(M * H * 128) / 256, 256, 0, stream>>>(k_h, T, M * H, 1.0f);

  // 4) retention (WMMA flash-style, causal decay)
  constexpr size_t ret_lds = (size_t)(64 * 256 + 64 * 256 + 512 * 64 + 64 * 64) * 2; // 136 KB
  retention_wmma<<<dim3(T / 64, H, B), 256, ret_lds, stream>>>(q_h, k_h, v_h, o_f32, T);

  // 5) RMS-norm + SiLU gate
  norm_gate<<<(M * H) / 8, 256, 0, stream>>>(o_f32, g_h, gnw, gat_h, M * H);

  // 6) output projection (WMMA GEMM, fp32 out)
  gemm_wmma<float><<<dim3(HID / 128, M / 128), 256, 0, stream>>>(gat_h, Wo_h, out, M, HID, NVG);
}
